// GATLayer_40862318854586
// MI455X (gfx1250) — matrix operand order repaired and checked by reading, not compiled, not run on hardware
//
#include <hip/hip_runtime.h>
#include <hip/hip_bf16.h>
#include <math.h>

typedef _Float16 v16h __attribute__((ext_vector_type(16)));
typedef _Float16 v8h  __attribute__((ext_vector_type(8)));
typedef _Float16 v4h  __attribute__((ext_vector_type(4)));
typedef float    v8f  __attribute__((ext_vector_type(8)));
typedef float    v4f  __attribute__((ext_vector_type(4)));

#define NNODES 8192
#define INDIM  256
#define OUTDIM 64

__device__ __forceinline__ float lrelu(float x) { return fmaxf(x, 0.2f * x); }

// ---------------------------------------------------------------------------
// Kernel 1: Wh = h @ W  (f32), store WhT in f16 [64][8192], e_src/e_dst (f32),
// and per-block max of e_dst. 128 blocks x 256 threads; 64 rows per block.
// ---------------------------------------------------------------------------
__global__ __launch_bounds__(256) void k1_project(
    const float* __restrict__ h, const float* __restrict__ W,
    const float* __restrict__ a_src, const float* __restrict__ a_dst,
    _Float16* __restrict__ WhT16, float* __restrict__ e_src,
    float* __restrict__ e_dst, float* __restrict__ pmax)
{
  __shared__ float WT[64][132];   // W transposed, K-half staged; pad 132 (16B-aligned rows)
  __shared__ float Wh[64][65];    // row-tile of Wh; pad 65 for column scans
  __shared__ float red[64];

  const int t = threadIdx.x;
  const int rowbase = blockIdx.x * 64;
  const int c = t & 63;
  const int rowsub = t >> 6;

  float accv[16];
#pragma unroll
  for (int i = 0; i < 16; ++i) accv[i] = 0.f;

  for (int half = 0; half < 2; ++half) {
    // stage WT[c][klocal] = W[(half*128+klocal)*64 + c]
#pragma unroll
    for (int i = 0; i < 8; ++i) {
      int u = t + i * 256;            // 0..2047
      int klocal = u >> 4;            // 0..127
      int cblk = (u & 15) * 4;
      v4f w4 = *(const v4f*)&W[(size_t)(half * 128 + klocal) * OUTDIM + cblk];
      WT[cblk + 0][klocal] = w4[0];
      WT[cblk + 1][klocal] = w4[1];
      WT[cblk + 2][klocal] = w4[2];
      WT[cblk + 3][klocal] = w4[3];
    }
    __syncthreads();
#pragma unroll 1
    for (int rr = 0; rr < 16; ++rr) {
      int row = rowsub + rr * 4;
      const v4f* hp = (const v4f*)&h[(size_t)(rowbase + row) * INDIM + half * 128];
      float a = accv[rr];
#pragma unroll 8
      for (int k4 = 0; k4 < 32; ++k4) {
        v4f hv = hp[k4];
        v4f wv = *(const v4f*)&WT[c][k4 * 4];
        a = fmaf(hv[0], wv[0], a);
        a = fmaf(hv[1], wv[1], a);
        a = fmaf(hv[2], wv[2], a);
        a = fmaf(hv[3], wv[3], a);
      }
      accv[rr] = a;
    }
    __syncthreads();
  }
#pragma unroll
  for (int rr = 0; rr < 16; ++rr) Wh[rowsub + rr * 4][c] = accv[rr];
  __syncthreads();

  // WhT16[c][row] = (f16) Wh[row][c]   (coalesced along row)
#pragma unroll
  for (int i = 0; i < 16; ++i) {
    int u = t + i * 256;               // 0..4095
    int c2 = u >> 6, r2 = u & 63;
    WhT16[(size_t)c2 * NNODES + rowbase + r2] = (_Float16)Wh[r2][c2];
  }
  if (t < 64) {
    float esv = 0.f, edv = 0.f;
    for (int cc = 0; cc < 64; ++cc) {
      float v = Wh[t][cc];
      esv = fmaf(v, a_src[cc], esv);
      edv = fmaf(v, a_dst[cc], edv);
    }
    e_src[rowbase + t] = esv;
    e_dst[rowbase + t] = edv;
    red[t] = edv;
  }
  __syncthreads();
  if (t == 0) {
    float m = red[0];
    for (int i = 1; i < 64; ++i) m = fmaxf(m, red[i]);
    pmax[blockIdx.x] = m;
  }
}

// Reduce 128 per-block maxima -> global max of e_dst (deterministic).
__global__ void k1b_gmax(const float* __restrict__ pmax, float* __restrict__ gmax)
{
  if (threadIdx.x == 0 && blockIdx.x == 0) {
    float m = pmax[0];
    for (int i = 1; i < 128; ++i) m = fmaxf(m, pmax[i]);
    *gmax = m;
  }
}

// ---------------------------------------------------------------------------
// Kernel 2: fused masked-softmax attention:  out = softmax(mask(leaky(e_src+e_dst))) @ Wh
// 256 blocks x 256 threads; block = 32 rows, full j sweep in BK=128 chunks.
// 8 waves per block: wave (rt,ct) owns one 16x16 f32 accumulator tile,
// fed by v_wmma_f32_16x16x32_f16 (p tile f16 x WhT tile f16).
// Fixed per-row max bound M_i = leaky(e_src[i] + max_j e_dst[j]) => no rescaling.
// ---------------------------------------------------------------------------
#define BK 128
#define PSTR 136   // padded LDS row stride (f16 elems): 272B, 16B-aligned, 2-way max conflict
#define WSTR 136

__global__ __launch_bounds__(256) void k2_gat(
    const float* __restrict__ adj, const _Float16* __restrict__ WhT16,
    const float* __restrict__ e_srcg, const float* __restrict__ e_dstg,
    const float* __restrict__ gmaxp, float* __restrict__ out)
{
  __shared__ __align__(16) _Float16 pbuf[2][32 * PSTR];   // p tile (f16), double buffered
  __shared__ __align__(16) _Float16 wbuf[2][64 * WSTR];   // WhT tile (f16), double buffered
  __shared__ float lparts[256];
  __shared__ float lrow[32];

  const int t = threadIdx.x;
  const int rowbase = blockIdx.x * 32;
  const float gmax = *gmaxp;

  // phase-A mapping: thread -> (row, 16-wide j slice)
  const int arowA = t >> 3;                 // 0..31
  const int j8 = (t & 7) * 16;              // 0,16,...,112
  const int rowg = rowbase + arowA;
  const float es = e_srcg[rowg];
  const float Mrow = lrelu(es + gmax);      // per-row score upper bound
  float lsum = 0.f;

  // wave mapping for WMMA
  const int lane = t & 31;
  const int w = t >> 5;
  const int rt = w >> 2;                    // row tile 0..1
  const int ct = w & 3;                     // col tile 0..3
  const int arow = rt * 16 + (lane & 15);
  const int klo  = (lane < 16) ? 0 : 8;     // A frag: K 0-7/16-23 vs 8-15/24-31
  const int bcol = ct * 16 + (lane & 15);
  const int bk0  = (lane < 16) ? 0 : 8;     // B frag: K 0-7/16-23 vs 8-15/24-31
  v8f acc{};

  v4f av[4], ev[4];
  v8h wreg[4];

  auto loadA = [&](int chunk) {
    const float* ap = &adj[(size_t)rowg * NNODES + chunk * BK + j8];
    const float* ep = &e_dstg[chunk * BK + j8];
#pragma unroll
    for (int i = 0; i < 4; ++i) {
      av[i] = __builtin_nontemporal_load((const v4f*)ap + i);  // stream adj, don't pollute caches
      ev[i] = *((const v4f*)ep + i);
    }
  };
  auto loadW = [&](int chunk) {
#pragma unroll
    for (int i = 0; i < 4; ++i) {
      int u = t * 4 + i;                    // 0..1023 16B-blocks of 64x128 f16 tile
      int cc = u >> 4, kb = u & 15;
      wreg[i] = *(const v8h*)&WhT16[(size_t)cc * NNODES + chunk * BK + kb * 8];
    }
  };
  auto storeW = [&](int bsel) {
#pragma unroll
    for (int i = 0; i < 4; ++i) {
      int u = t * 4 + i;
      int cc = u >> 4, kb = u & 15;
      *(v8h*)&wbuf[bsel][cc * WSTR + kb * 8] = wreg[i];
    }
  };
  auto storeA = [&](int chunk, int bsel) {
    _Float16* pd = &pbuf[bsel][arowA * PSTR + j8];
#pragma unroll
    for (int i = 0; i < 4; ++i) {
      int jg = chunk * BK + j8 + i * 4;
      v4h ph;
#pragma unroll
      for (int u = 0; u < 4; ++u) {
        float x = es + ev[i][u];
        float s = fmaxf(x, 0.2f * x);                 // leaky_relu
        float p = __expf(s - Mrow);                   // <= 1, no overflow
        bool ok = (av[i][u] != 0.f) || ((jg + u) == rowg);  // adj edge or self-loop
        p = ok ? p : 0.f;
        lsum += p;
        ph[u] = (_Float16)p;
      }
      *(v4h*)(pd + i * 4) = ph;
    }
  };
  auto mma = [&](int bsel) {
    const _Float16* pb = &pbuf[bsel][arow * PSTR];
    const _Float16* wb = &wbuf[bsel][bcol * WSTR];
#pragma unroll
    for (int kb = 0; kb < BK; kb += 32) {
      v8h alo = *(const v8h*)(pb + kb + klo);         // A: K klo..klo+7
      v8h ahi = *(const v8h*)(pb + kb + klo + 16);    // A: K klo+16..klo+23
      v8h blo = *(const v8h*)(wb + kb + bk0);         // B: K bk0..bk0+7
      v8h bhi = *(const v8h*)(wb + kb + bk0 + 16);    // B: K bk0+16..bk0+23
      v16h A = __builtin_shufflevector(alo, ahi, 0,1,2,3,4,5,6,7,8,9,10,11,12,13,14,15);
      v16h B = __builtin_shufflevector(blo, bhi, 0,1,2,3,4,5,6,7,8,9,10,11,12,13,14,15);
      acc = __builtin_amdgcn_wmma_f32_16x16x32_f16(false, A, false, B, (short)0, acc,
                                                   false, false);
    }
  };

  // prologue: fill buffer 0
  loadA(0); loadW(0);
  storeW(0); storeA(0, 0);
  __syncthreads();

  for (int chunk = 0; chunk < 64; ++chunk) {
    int cur = chunk & 1;
    if (chunk + 1 < 64) { loadA(chunk + 1); loadW(chunk + 1); }  // issue global loads first
    mma(cur);                                                    // overlap WMMA with loads
    if (chunk + 1 < 64) { storeW(cur ^ 1); storeA(chunk + 1, cur ^ 1); }
    __syncthreads();
  }

  // deterministic l reduction: 8 partials per row
  lparts[t] = lsum;
  __syncthreads();
  if (t < 32) {
    float s = 0.f;
    for (int i = 0; i < 8; ++i) s += lparts[t * 8 + i];
    lrow[t] = s;
  }
  __syncthreads();

  // C layout: VGPR r holds rows (r, r+8) for lane halves; cols = lane&15
#pragma unroll
  for (int r = 0; r < 8; ++r) {
    int rowl = rt * 16 + r + ((lane >= 16) ? 8 : 0);
    out[(size_t)(rowbase + rowl) * OUTDIM + ct * 16 + (lane & 15)] = acc[r] / lrow[rowl];
  }
}

// ---------------------------------------------------------------------------
extern "C" void kernel_launch(void* const* d_in, const int* in_sizes, int n_in,
                              void* d_out, int out_size, void* d_ws, size_t ws_size,
                              hipStream_t stream)
{
  (void)in_sizes; (void)n_in; (void)out_size; (void)ws_size;
  const float* h     = (const float*)d_in[0];
  const float* adj   = (const float*)d_in[1];
  const float* W     = (const float*)d_in[2];
  const float* a_src = (const float*)d_in[3];
  const float* a_dst = (const float*)d_in[4];
  float* out = (float*)d_out;

  char* ws = (char*)d_ws;
  _Float16* WhT16 = (_Float16*)ws;                       // 64*8192*2 = 1 MB
  float* e_src = (float*)(ws + 1048576);                 // 32 KB
  float* e_dst = (float*)(ws + 1048576 + 32768);         // 32 KB
  float* pmax  = (float*)(ws + 1048576 + 65536);         // 512 B
  float* gmax  = (float*)(ws + 1048576 + 65536 + 512);   // 4 B

  k1_project<<<128, 256, 0, stream>>>(h, W, a_src, a_dst, WhT16, e_src, e_dst, pmax);
  k1b_gmax<<<1, 32, 0, stream>>>(pmax, gmax);
  k2_gat<<<256, 256, 0, stream>>>(adj, WhT16, e_src, e_dst, gmax, out);
}
